// GraphTransformer_16174846837043
// MI455X (gfx1250) — matrix operand order repaired and checked by reading, not compiled, not run on hardware
//
#include <hip/hip_runtime.h>

// ---------------------------------------------------------------------------
// GraphTransformer (TransformerConv x3) for MI455X / gfx1250, wave32 + WMMA.
// Dense GEMMs use v_wmma_f32_16x16x32_bf16 (1 wave = 16x64 output tile,
// double-buffered K loop); edge softmax/scatter uses native f32 global
// atomics; graph-LayerNorm via two-pass reduction.
// ---------------------------------------------------------------------------

#define DEVFN __device__ __forceinline__

typedef unsigned short u16;
typedef unsigned int   u32;
typedef __attribute__((ext_vector_type(16))) __bf16 v16bf;
typedef __attribute__((ext_vector_type(8)))  float  v8f;

static constexpr int N_NODES = 30000;
static constexpr int N_EDGES = 480000;
static constexpr int D_IN    = 128;
static constexpr int D_HID   = 256;   // HEADS * HID
static constexpr int HEADS   = 4;
static constexpr int HID     = 64;
static constexpr int E_EMB   = 32;

// ---- helpers --------------------------------------------------------------
DEVFN u16 f2bf(float f) {
    union { float f; u32 u; } x; x.f = f;
    u32 r = x.u + 0x7FFFu + ((x.u >> 16) & 1u);   // round-to-nearest-even
    return (u16)(r >> 16);
}
DEVFN float bf2f(u32 lo16) {
    union { u32 u; float f; } x; x.u = lo16 << 16;
    return x.f;
}
// order-preserving float<->uint map for atomicMax on floats
DEVFN u32 f2ord(float f) {
    u32 u = __float_as_uint(f);
    return (u & 0x80000000u) ? ~u : (u | 0x80000000u);
}
DEVFN float ord2f(u32 u) {
    return __uint_as_float((u & 0x80000000u) ? (u & 0x7FFFFFFFu) : ~u);
}

union ABfrag { uint4 u[2]; v16bf v; };

// A lane (ISA 7.12.2): m = lane&15, holds k in [k0+8g,+8) U [k0+16+8g,+8)
DEVFN void load_afrag(ABfrag& f, const u16* __restrict__ arow, int k0, int g) {
    f.u[0] = *(const uint4*)(arow + k0 + 8 * g);
    f.u[1] = *(const uint4*)(arow + k0 + 16 + 8 * g);
}
// B lane: col n = lane&15, holds k in [k0+8g,+8) U [k0+16+8g,+8)
DEVFN void load_bfrag(ABfrag& f, const u16* __restrict__ brow, int k0, int g) {
    f.u[0] = *(const uint4*)(brow + k0 + 8 * g);
    f.u[1] = *(const uint4*)(brow + k0 + 16 + 8 * g);
}
DEVFN v8f wmma_bf16(const ABfrag& a, const ABfrag& b, v8f c) {
    return __builtin_amdgcn_wmma_f32_16x16x32_bf16(
        false, a.v, false, b.v, (short)0, c, false, false);
}

// ---------------------------------------------------------------------------
// bf16 WMMA GEMM: Y[rows x OUT] = A[rows x K] * W[K x OUT] (+bias).
// A: bf16 row-major (stride K). Wt: bf16 transposed weights [OUT x K].
// One wave computes a 16x64 tile (4 accumulators, shared A fragment);
// K loop is software double-buffered so loads for step k+1 are in flight
// while the 4 WMMAs of step k execute.
// grid = (rows/16, ceil(OUT/256)), block = 128 (4 waves -> 16 x 256 tile).
// ---------------------------------------------------------------------------
__global__ __launch_bounds__(128)
void gemm_bf16_wmma(const u16* __restrict__ A, const u16* __restrict__ Wt,
                    const float* __restrict__ bias,
                    float* __restrict__ Yf, u16* __restrict__ Yh,
                    int K, int OUT, int relu)
{
    const int lane = threadIdx.x & 31;
    const int wave = threadIdx.x >> 5;
    const int g    = lane >> 4;       // half-wave group
    const int ln16 = lane & 15;
    const int rowBase = blockIdx.x * 16;
    const int colBase = blockIdx.y * 256 + wave * 64;
    if (colBase >= OUT) return;

    const u16* arow = A + (size_t)(rowBase + ln16) * K;
    const u16* brow0 = Wt + (size_t)(colBase + 0  + ln16) * K;
    const u16* brow1 = Wt + (size_t)(colBase + 16 + ln16) * K;
    const u16* brow2 = Wt + (size_t)(colBase + 32 + ln16) * K;
    const u16* brow3 = Wt + (size_t)(colBase + 48 + ln16) * K;

    v8f acc0 = {0.f,0.f,0.f,0.f,0.f,0.f,0.f,0.f};
    v8f acc1 = acc0, acc2 = acc0, acc3 = acc0;

    ABfrag ac, b0c, b1c, b2c, b3c;
    load_afrag(ac, arow, 0, g);
    load_bfrag(b0c, brow0, 0, g);
    load_bfrag(b1c, brow1, 0, g);
    load_bfrag(b2c, brow2, 0, g);
    load_bfrag(b3c, brow3, 0, g);

    int k0 = 0;
    for (; k0 + 32 < K; k0 += 32) {
        ABfrag an, b0n, b1n, b2n, b3n;
        const int k1 = k0 + 32;
        load_afrag(an, arow, k1, g);
        load_bfrag(b0n, brow0, k1, g);
        load_bfrag(b1n, brow1, k1, g);
        load_bfrag(b2n, brow2, k1, g);
        load_bfrag(b3n, brow3, k1, g);
        if (k1 + 32 < K) {
            __builtin_prefetch(arow + k1 + 32, 0, 3);
            __builtin_prefetch(brow0 + k1 + 32, 0, 3);
            __builtin_prefetch(brow2 + k1 + 32, 0, 3);
        }
        acc0 = wmma_bf16(ac, b0c, acc0);
        acc1 = wmma_bf16(ac, b1c, acc1);
        acc2 = wmma_bf16(ac, b2c, acc2);
        acc3 = wmma_bf16(ac, b3c, acc3);
        ac = an; b0c = b0n; b1c = b1n; b2c = b2n; b3c = b3n;
    }
    acc0 = wmma_bf16(ac, b0c, acc0);
    acc1 = wmma_bf16(ac, b1c, acc1);
    acc2 = wmma_bf16(ac, b2c, acc2);
    acc3 = wmma_bf16(ac, b3c, acc3);

    // epilogue: D row = rowBase + r + 8*g, cols colBase + t*16 + ln16
    #pragma unroll
    for (int t = 0; t < 4; ++t) {
        const v8f acc = (t == 0) ? acc0 : (t == 1) ? acc1 : (t == 2) ? acc2 : acc3;
        const int col = colBase + t * 16 + ln16;
        const float bcol = bias ? bias[col] : 0.f;
        #pragma unroll
        for (int r = 0; r < 8; ++r) {
            float y = acc[r] + bcol;
            if (relu) y = fmaxf(y, 0.f);
            size_t offY = (size_t)(rowBase + r + 8 * g) * OUT + col;
            if (Yf) Yf[offY] = y;
            if (Yh) Yh[offY] = f2bf(y);
        }
    }
}

// ---- weight convert: W[K x OUT] f32 -> Wt[OUT x K] bf16 -------------------
__global__ __launch_bounds__(256)
void convert_wt_kernel(const float* __restrict__ W, u16* __restrict__ Wt,
                       int K, int OUT)
{
    int idx = blockIdx.x * 256 + threadIdx.x;
    if (idx >= K * OUT) return;
    int k = idx / OUT, o = idx % OUT;
    Wt[(size_t)o * K + k] = f2bf(W[idx]);
}

// ---- f32 -> bf16 elementwise ----------------------------------------------
__global__ __launch_bounds__(256)
void convert_f32_bf16_kernel(const float* __restrict__ x, u16* __restrict__ xb,
                             int n)
{
    int idx = blockIdx.x * 256 + threadIdx.x;
    if (idx < n) xb[idx] = f2bf(x[idx]);
}

// ---- fused edge MLP: edge_attr[E x 16] -> relu(W1)+W2 -> e_emb bf16 [E x 32]
__global__ __launch_bounds__(256)
void edge_mlp_kernel(const float* __restrict__ ea,
                     const float* __restrict__ W1, const float* __restrict__ b1,
                     const float* __restrict__ W2, const float* __restrict__ b2,
                     u16* __restrict__ emb)
{
    __shared__ float sW1[16 * 64], sb1[64], sW2[64 * 32], sb2[32];
    for (int i = threadIdx.x; i < 16 * 64; i += 256) sW1[i] = W1[i];
    for (int i = threadIdx.x; i < 64;      i += 256) sb1[i] = b1[i];
    for (int i = threadIdx.x; i < 64 * 32; i += 256) sW2[i] = W2[i];
    if (threadIdx.x < 32) sb2[threadIdx.x] = b2[threadIdx.x];
    __syncthreads();

    int e = blockIdx.x * 256 + threadIdx.x;
    if (e >= N_EDGES) return;

    float in[16];
    #pragma unroll
    for (int i = 0; i < 16; ++i) in[i] = ea[(size_t)e * 16 + i];

    float out[32];
    #pragma unroll
    for (int o = 0; o < 32; ++o) out[o] = sb2[o];

    for (int j = 0; j < 64; ++j) {
        float acc = sb1[j];
        #pragma unroll
        for (int i = 0; i < 16; ++i) acc += in[i] * sW1[i * 64 + j];
        acc = fmaxf(acc, 0.f);
        #pragma unroll
        for (int o = 0; o < 32; ++o) out[o] += acc * sW2[j * 32 + o];
    }
    #pragma unroll
    for (int o = 0; o < 32; ++o) emb[(size_t)e * 32 + o] = f2bf(out[o]);
}

// ---- attention pass 1: alpha + segment max --------------------------------
__global__ __launch_bounds__(256)
void attn_alpha_kernel(const int* __restrict__ ei,
                       const float* __restrict__ Q, const float* __restrict__ Kf,
                       const u16* __restrict__ eproj,
                       float* __restrict__ alpha, u32* __restrict__ mu)
{
    int idx = blockIdx.x * 256 + threadIdx.x;      // (edge, head)
    if (idx >= N_EDGES * HEADS) return;
    int e = idx >> 2, h = idx & 3;
    int src = ei[e], dst = ei[N_EDGES + e];

    const float4* qp = (const float4*)(Q  + (size_t)dst * D_HID + h * HID);
    const float4* kp = (const float4*)(Kf + (size_t)src * D_HID + h * HID);
    const u16*    ep = eproj + (size_t)e * D_HID + h * HID;

    float acc = 0.f;
    #pragma unroll 4
    for (int i = 0; i < 16; ++i) {
        float4 q = qp[i], k = kp[i];
        uint2 eu = *(const uint2*)(ep + i * 4);
        acc += q.x * (k.x + bf2f(eu.x & 0xFFFFu));
        acc += q.y * (k.y + bf2f(eu.x >> 16));
        acc += q.z * (k.z + bf2f(eu.y & 0xFFFFu));
        acc += q.w * (k.w + bf2f(eu.y >> 16));
    }
    acc *= 0.125f;                                  // 1/sqrt(64)
    alpha[idx] = acc;
    atomicMax(mu + (size_t)dst * HEADS + h, f2ord(acc));
}

// ---- attention pass 2: exp + segment sum ----------------------------------
__global__ __launch_bounds__(256)
void attn_expsum_kernel(const int* __restrict__ ei,
                        const float* __restrict__ alpha, const u32* __restrict__ mu,
                        float* __restrict__ aexp, float* __restrict__ s)
{
    int idx = blockIdx.x * 256 + threadIdx.x;
    if (idx >= N_EDGES * HEADS) return;
    int e = idx >> 2, h = idx & 3;
    int dst = ei[N_EDGES + e];
    float m = ord2f(mu[(size_t)dst * HEADS + h]);
    float a = __expf(alpha[idx] - m);
    aexp[idx] = a;
    atomicAdd(s + (size_t)dst * HEADS + h, a);
}

// ---- attention pass 3: scatter attn * (v[src] + eproj) --------------------
__global__ __launch_bounds__(256)
void attn_scatter_kernel(const int* __restrict__ ei,
                         const float* __restrict__ V, const u16* __restrict__ eproj,
                         const float* __restrict__ aexp, const float* __restrict__ s,
                         float* __restrict__ out)
{
    int tid = blockIdx.x * 256 + threadIdx.x;      // E * 64 threads, 4ch each
    if (tid >= N_EDGES * 64) return;
    int e = tid >> 6;
    int t = tid & 63;
    int ch = t << 2;
    int h  = t >> 4;
    int src = ei[e], dst = ei[N_EDGES + e];

    float sv   = s[(size_t)dst * HEADS + h];
    float attn = aexp[(size_t)e * HEADS + h] / fmaxf(sv, 1e-16f);

    float4 v = *(const float4*)(V + (size_t)src * D_HID + ch);
    uint2 eu = *(const uint2*)(eproj + (size_t)e * D_HID + ch);
    float* o = out + (size_t)dst * D_HID + ch;
    atomicAdd(o + 0, attn * (v.x + bf2f(eu.x & 0xFFFFu)));
    atomicAdd(o + 1, attn * (v.y + bf2f(eu.x >> 16)));
    atomicAdd(o + 2, attn * (v.z + bf2f(eu.y & 0xFFFFu)));
    atomicAdd(o + 3, attn * (v.w + bf2f(eu.y >> 16)));
}

// ---- graph layernorm: whole-tensor sum / sumsq ----------------------------
__global__ __launch_bounds__(256)
void ln_stats_kernel(const float* __restrict__ x, float* __restrict__ sums)
{
    __shared__ float s1[256], s2[256];
    int idx = blockIdx.x * 256 + threadIdx.x;
    float v = x[idx];
    s1[threadIdx.x] = v;
    s2[threadIdx.x] = v * v;
    __syncthreads();
    for (int st = 128; st > 0; st >>= 1) {
        if (threadIdx.x < st) {
            s1[threadIdx.x] += s1[threadIdx.x + st];
            s2[threadIdx.x] += s2[threadIdx.x + st];
        }
        __syncthreads();
    }
    if (threadIdx.x == 0) {
        atomicAdd(&sums[0], s1[0]);
        atomicAdd(&sums[1], s2[0]);
    }
}

// ---- graph layernorm apply + relu + bf16 convert (feeds next WMMA GEMM) ---
__global__ __launch_bounds__(256)
void ln_apply_kernel(const float* __restrict__ x, const float* __restrict__ sums,
                     const float* __restrict__ w, const float* __restrict__ b,
                     u16* __restrict__ xb)
{
    int idx = blockIdx.x * 256 + threadIdx.x;      // N*256 elements
    const float inv = 1.f / (float)(N_NODES * D_HID);
    float mean = sums[0] * inv;
    float var  = fmaxf(sums[1] * inv - mean * mean, 0.f);
    float den  = __fsqrt_rn(var) + 1e-5f;
    int ch = idx & (D_HID - 1);
    float y = (x[idx] - mean) / den * w[ch] + b[ch];
    xb[idx] = f2bf(fmaxf(y, 0.f));
}

// ---- final linear 128 -> 2 (tiny; scalar with weights in LDS) -------------
__global__ __launch_bounds__(256)
void final_mlp_kernel(const u16* __restrict__ Hb,
                      const float* __restrict__ W, const float* __restrict__ b,
                      float* __restrict__ out)
{
    __shared__ float sW[256], sb[2];
    for (int i = threadIdx.x; i < 256; i += 256) sW[i] = W[i];
    if (threadIdx.x < 2) sb[threadIdx.x] = b[threadIdx.x];
    __syncthreads();
    int n = blockIdx.x * 256 + threadIdx.x;
    if (n >= N_NODES) return;
    const u16* hp = Hb + (size_t)n * 128;
    float a0 = sb[0], a1 = sb[1];
    for (int k = 0; k < 128; k += 2) {
        u32 hu = *(const u32*)(hp + k);
        float h0 = bf2f(hu & 0xFFFFu), h1 = bf2f(hu >> 16);
        a0 += h0 * sW[k * 2 + 0] + h1 * sW[(k + 1) * 2 + 0];
        a1 += h0 * sW[k * 2 + 1] + h1 * sW[(k + 1) * 2 + 1];
    }
    out[(size_t)n * 2 + 0] = a0;
    out[(size_t)n * 2 + 1] = a1;
}

// ---------------------------------------------------------------------------
// Host launch.  Input flattening (setup_inputs dict insertion order, None
// biases dropped):
//   0 x | 1 edge_index | 2 edge_attr | 3..6 edge_mlp {W,b,W,b}
//   7 + 11*l : layer l = qW qb kW kb vW vb eW skipW skipb lnw lnb
//   40..43  : mlp0.W mlp0.b mlp1.W mlp1.b
// ---------------------------------------------------------------------------
extern "C" void kernel_launch(void* const* d_in, const int* in_sizes, int n_in,
                              void* d_out, int out_size, void* d_ws, size_t ws_size,
                              hipStream_t stream)
{
    (void)in_sizes; (void)n_in; (void)out_size; (void)ws_size;

    const float* x_in  = (const float*)d_in[0];
    const int*   ei    = (const int*)d_in[1];
    const float* eattr = (const float*)d_in[2];
    const float* em0W  = (const float*)d_in[3];
    const float* em0b  = (const float*)d_in[4];
    const float* em1W  = (const float*)d_in[5];
    const float* em1b  = (const float*)d_in[6];
    const float* mlp0W = (const float*)d_in[40];
    const float* mlp0b = (const float*)d_in[41];
    const float* mlp1W = (const float*)d_in[42];
    const float* mlp1b = (const float*)d_in[43];
    float* out = (float*)d_out;

    // ---- carve workspace ----
    char* wsp = (char*)d_ws;
    size_t off = 0;
    auto ALLOC = [&](size_t bytes) -> char* {
        char* p = wsp + off;
        off += (bytes + 255) & ~(size_t)255;
        return p;
    };
    u16*   Xb    = (u16*)  ALLOC((size_t)N_NODES * D_HID * 2);   // bf16 activations
    u16*   Hb    = (u16*)  ALLOC((size_t)N_NODES * 128 * 2);     // mlp0 output
    float* Qb    = (float*)ALLOC((size_t)N_NODES * D_HID * 4);
    float* Kb    = (float*)ALLOC((size_t)N_NODES * D_HID * 4);
    float* Vb    = (float*)ALLOC((size_t)N_NODES * D_HID * 4);
    float* OUTb  = (float*)ALLOC((size_t)N_NODES * D_HID * 4);   // skip + attn accum
    u16*   EEMB  = (u16*)  ALLOC((size_t)N_EDGES * E_EMB * 2);
    u16*   EPROJ = (u16*)  ALLOC((size_t)N_EDGES * D_HID * 2);
    float* ALPHA = (float*)ALLOC((size_t)N_EDGES * HEADS * 4);
    float* AEXP  = (float*)ALLOC((size_t)N_EDGES * HEADS * 4);
    u32*   MU    = (u32*)  ALLOC((size_t)N_NODES * HEADS * 4);
    float* SS    = (float*)ALLOC((size_t)N_NODES * HEADS * 4);
    float* SUMS  = (float*)ALLOC(2 * 4);
    u16*   WTQ   = (u16*)  ALLOC((size_t)D_HID * D_HID * 2);
    u16*   WTK   = (u16*)  ALLOC((size_t)D_HID * D_HID * 2);
    u16*   WTV   = (u16*)  ALLOC((size_t)D_HID * D_HID * 2);
    u16*   WTS   = (u16*)  ALLOC((size_t)D_HID * D_HID * 2);
    u16*   WTE   = (u16*)  ALLOC((size_t)E_EMB * D_HID * 2);
    u16*   WTM   = (u16*)  ALLOC((size_t)D_HID * 128 * 2);

    const dim3 B256(256), B128(128);

    // initial activations -> bf16
    convert_f32_bf16_kernel<<<(N_NODES * D_IN + 255) / 256, B256, 0, stream>>>(
        x_in, Xb, N_NODES * D_IN);

    // edge embedding MLP -> bf16 [E x 32]
    edge_mlp_kernel<<<(N_EDGES + 255) / 256, B256, 0, stream>>>(
        eattr, em0W, em0b, em1W, em1b, EEMB);

    for (int l = 0; l < 3; ++l) {
        const int Din  = (l == 0) ? D_IN : D_HID;
        const int base = 7 + 11 * l;
        const float* qW = (const float*)d_in[base + 0];
        const float* qb = (const float*)d_in[base + 1];
        const float* kW = (const float*)d_in[base + 2];
        const float* kb = (const float*)d_in[base + 3];
        const float* vW = (const float*)d_in[base + 4];
        const float* vb = (const float*)d_in[base + 5];
        const float* eW = (const float*)d_in[base + 6];
        const float* sW = (const float*)d_in[base + 7];
        const float* sb = (const float*)d_in[base + 8];
        const float* lnw = (const float*)d_in[base + 9];
        const float* lnb = (const float*)d_in[base + 10];

        // weight transpose + bf16 convert
        int nw = Din * D_HID;
        convert_wt_kernel<<<(nw + 255) / 256, B256, 0, stream>>>(qW, WTQ, Din, D_HID);
        convert_wt_kernel<<<(nw + 255) / 256, B256, 0, stream>>>(kW, WTK, Din, D_HID);
        convert_wt_kernel<<<(nw + 255) / 256, B256, 0, stream>>>(vW, WTV, Din, D_HID);
        convert_wt_kernel<<<(nw + 255) / 256, B256, 0, stream>>>(sW, WTS, Din, D_HID);
        convert_wt_kernel<<<(E_EMB * D_HID + 255) / 256, B256, 0, stream>>>(
            eW, WTE, E_EMB, D_HID);

        // node GEMMs (WMMA): Q, K, V, and OUT = skip(x)
        dim3 gN(N_NODES / 16, (D_HID + 255) / 256);
        gemm_bf16_wmma<<<gN, B128, 0, stream>>>(Xb, WTQ, qb, Qb,   nullptr, Din, D_HID, 0);
        gemm_bf16_wmma<<<gN, B128, 0, stream>>>(Xb, WTK, kb, Kb,   nullptr, Din, D_HID, 0);
        gemm_bf16_wmma<<<gN, B128, 0, stream>>>(Xb, WTV, vb, Vb,   nullptr, Din, D_HID, 0);
        gemm_bf16_wmma<<<gN, B128, 0, stream>>>(Xb, WTS, sb, OUTb, nullptr, Din, D_HID, 0);

        // edge projection GEMM (WMMA, K=32): EPROJ = e_emb @ We  (bf16 out)
        dim3 gE(N_EDGES / 16, (D_HID + 255) / 256);
        gemm_bf16_wmma<<<gE, B128, 0, stream>>>(EEMB, WTE, nullptr, nullptr, EPROJ,
                                                E_EMB, D_HID, 0);

        // segment softmax state
        hipMemsetAsync(MU,   0, (size_t)N_NODES * HEADS * 4, stream);
        hipMemsetAsync(SS,   0, (size_t)N_NODES * HEADS * 4, stream);
        hipMemsetAsync(SUMS, 0, 2 * 4, stream);

        attn_alpha_kernel<<<(N_EDGES * HEADS + 255) / 256, B256, 0, stream>>>(
            ei, Qb, Kb, EPROJ, ALPHA, MU);
        attn_expsum_kernel<<<(N_EDGES * HEADS + 255) / 256, B256, 0, stream>>>(
            ei, ALPHA, MU, AEXP, SS);
        attn_scatter_kernel<<<(N_EDGES * 64 + 255) / 256, B256, 0, stream>>>(
            ei, Vb, EPROJ, AEXP, SS, OUTb);

        // graph layernorm + relu -> bf16 activations for next layer
        ln_stats_kernel<<<N_NODES * D_HID / 256, B256, 0, stream>>>(OUTb, SUMS);
        ln_apply_kernel<<<N_NODES * D_HID / 256, B256, 0, stream>>>(
            OUTb, SUMS, lnw, lnb, Xb);
    }

    // final MLP: 256 -> 128 (WMMA + relu, bf16 out) -> 2 (scalar)
    convert_wt_kernel<<<(D_HID * 128 + 255) / 256, B256, 0, stream>>>(
        mlp0W, WTM, D_HID, 128);
    dim3 gM(N_NODES / 16, 1);
    gemm_bf16_wmma<<<gM, B128, 0, stream>>>(Xb, WTM, mlp0b, nullptr, Hb, D_HID, 128, 1);
    final_mlp_kernel<<<(N_NODES + 255) / 256, B256, 0, stream>>>(Hb, mlp1W, mlp1b, out);
}
